// PMWA_3676492005787
// MI455X (gfx1250) — matrix operand order repaired and checked by reading, not compiled, not run on hardware
//
#include <hip/hip_runtime.h>
#include <math.h>

typedef __attribute__((ext_vector_type(16))) _Float16 v16h;
typedef __attribute__((ext_vector_type(8)))  _Float16 v8h;
typedef __attribute__((ext_vector_type(4)))  _Float16 v4h;
typedef __attribute__((ext_vector_type(8)))  float    v8f;

#define DIM       128
#define NUM_HOPS  3
#define SIGMA     0.1f
#define EPS       1e-12f

// ---------------------------------------------------------------------------
// Kernel 1: row-wise L2 normalize; writes f32 row (output) + f16 shadow row
// used by the WMMA attention kernel. One wave32 per row, 4 floats per lane.
// ---------------------------------------------------------------------------
__global__ void pmwa_norm_kernel(const float* __restrict__ in,
                                 float* __restrict__ out,
                                 _Float16* __restrict__ out16, int n) {
    int wave = (int)((blockIdx.x * (unsigned)blockDim.x + threadIdx.x) >> 5);
    int lane = threadIdx.x & 31;
    if (wave >= n) return;
    const float4 v = ((const float4*)(in + (size_t)wave * DIM))[lane];
    float ss = v.x * v.x + v.y * v.y + v.z * v.z + v.w * v.w;
    #pragma unroll
    for (int off = 16; off; off >>= 1) ss += __shfl_xor(ss, off, 32);
    float inv = 1.0f / fmaxf(sqrtf(ss), EPS);
    float4 o = {v.x * inv, v.y * inv, v.z * inv, v.w * inv};
    ((float4*)(out + (size_t)wave * DIM))[lane] = o;
    v4h o16 = {(_Float16)o.x, (_Float16)o.y, (_Float16)o.z, (_Float16)o.w};
    *(v4h*)(out16 + (size_t)wave * DIM + lane * 4) = o16;
}

// ---------------------------------------------------------------------------
// Kernel 2: per-edge attention via WMMA, pure-f16 fragment loads.
// One wave handles 16 edges: A = 16 src rows (16xK), B = 16 dst rows as
// columns (Kx16); C = Gram tile, diagonal = edge dots; sigmoid -> alpha[e].
// Fragment layouts are contiguous per lane:
//   A (16x32 f16): lane holds M=lane&15; halves K[kh..kh+7], K[kh+8..kh+15],
//                  kh = (lane>=16) ? 16 : 0 -> two 16B loads
//   B (32x16 f16): lane holds N=lane&15; halves K[kh..kh+15],
//                  kh = (lane>=16) ? 16 : 0 -> one 32B load
// ---------------------------------------------------------------------------
__global__ void pmwa_alpha_wmma_kernel(const _Float16* __restrict__ h16,
                                       const int* __restrict__ src,
                                       const int* __restrict__ dst,
                                       float* __restrict__ alpha, int E) {
    int wave = (int)((blockIdx.x * (unsigned)blockDim.x + threadIdx.x) >> 5);
    int lane = threadIdx.x & 31;
    int e0 = wave * 16;
    if (e0 >= E) return;                 // wave-uniform; EXEC stays all-ones

    int m   = lane & 15;                 // A row / B column this lane feeds
    int em  = e0 + m;
    int eok = (em < E) ? em : (E - 1);   // clamp tail, keep all lanes active
    const _Float16* pS = h16 + (size_t)src[eok] * DIM;
    const _Float16* pD = h16 + (size_t)dst[eok] * DIM;

    int hi     = lane >> 4;
    int khalfA = hi ? 16 : 0;
    int khalfB = hi ? 16 : 0;

    v8f c = {};
    #pragma unroll
    for (int kb = 0; kb < 4; ++kb) {     // K chunks of 32, K = 128 total
        int kbase = kb * 32;
        v8h alo = *(const v8h*)(pS + kbase + khalfA);
        v8h ahi = *(const v8h*)(pS + kbase + khalfA + 8);
        v16h a = __builtin_shufflevector(alo, ahi,
                     0, 1, 2, 3, 4, 5, 6, 7, 8, 9, 10, 11, 12, 13, 14, 15);
        v16h b = *(const v16h*)(pD + kbase + khalfB);
        c = __builtin_amdgcn_wmma_f32_16x16x32_f16(
                /*neg_a=*/false, a, /*neg_b=*/false, b,
                /*c_mod=*/(short)0, c, /*reuse_a=*/false, /*reuse_b=*/false);
    }

    // Diagonal: lane l holds (M = r + 8*hi, N = l&15) in c[r];
    // M==N lives in lanes 0-7 (r = lane) and lanes 24-31 (r = lane-24).
    int n = lane & 15;
    int r = n - (hi ? 8 : 0);
    if (r >= 0 && r < 8) {
        float dotv = 0.0f;
        #pragma unroll
        for (int i = 0; i < 8; ++i)
            if (i == r) dotv = c[i];
        int e = e0 + n;
        if (e < E)
            alpha[e] = 1.0f / (1.0f + __expf(-dotv));
    }
}

// ---------------------------------------------------------------------------
// Kernel 3: scatter-add  aggr[dst[e]] += alpha[e] * h[src[e]]  (f32 path)
// One wave per edge, float4 per lane -> global_atomic_add_f32 (L2-resident).
// ---------------------------------------------------------------------------
__global__ void pmwa_scatter_kernel(const float* __restrict__ h,
                                    const int* __restrict__ src,
                                    const int* __restrict__ dst,
                                    const float* __restrict__ alpha,
                                    float* __restrict__ aggr, int E) {
    long long tid = blockIdx.x * (long long)blockDim.x + threadIdx.x;
    int e = (int)(tid >> 5);
    if (e >= E) return;
    int j = (int)(tid & 31) * 4;
    float a = alpha[e];
    int s = src[e], d = dst[e];
    const float4 v = *(const float4*)(h + (size_t)s * DIM + j);
    float* o = aggr + (size_t)d * DIM + j;
    atomicAdd(o + 0, a * v.x);
    atomicAdd(o + 1, a * v.y);
    atomicAdd(o + 2, a * v.z);
    atomicAdd(o + 3, a * v.w);
}

// ---------------------------------------------------------------------------
// Kernel 4: h_next = l2_normalize(aggr + sigma * noise); also emits the f16
// shadow copy for the next hop's WMMA pass.
// Deterministic counter-hash Box-Muller stand-in for jax.random.normal.
// ---------------------------------------------------------------------------
__device__ __forceinline__ unsigned pmwa_pcg(unsigned x) {
    x = x * 747796405u + 2891336453u;
    unsigned w = ((x >> ((x >> 28) + 4u)) ^ x) * 277803737u;
    return (w >> 22) ^ w;
}
__device__ __forceinline__ float pmwa_gauss(unsigned s) {
    unsigned u1 = pmwa_pcg(s);
    unsigned u2 = pmwa_pcg(s ^ 0x9E3779B9u);
    float f1 = ((float)u1 + 1.0f) * 2.3283064e-10f;   // (0,1]
    float f2 = (float)u2 * 2.3283064e-10f;            // [0,1)
    return sqrtf(-2.0f * __logf(f1)) * __cosf(6.2831853f * f2);
}

__global__ void pmwa_finalize_kernel(const float* __restrict__ aggr,
                                     float* __restrict__ out,
                                     _Float16* __restrict__ out16,
                                     int n, int hop) {
    int wave = (int)((blockIdx.x * (unsigned)blockDim.x + threadIdx.x) >> 5);
    int lane = threadIdx.x & 31;
    if (wave >= n) return;
    float4 v = ((const float4*)(aggr + (size_t)wave * DIM))[lane];
    int col = lane * 4;
    unsigned base = (unsigned)(hop + 1) * 0x9E3779B1u ^ (unsigned)wave * 0x85EBCA77u;
    v.x += SIGMA * pmwa_gauss(base ^ (unsigned)(col + 0) * 0xC2B2AE3Du);
    v.y += SIGMA * pmwa_gauss(base ^ (unsigned)(col + 1) * 0xC2B2AE3Du);
    v.z += SIGMA * pmwa_gauss(base ^ (unsigned)(col + 2) * 0xC2B2AE3Du);
    v.w += SIGMA * pmwa_gauss(base ^ (unsigned)(col + 3) * 0xC2B2AE3Du);
    float ss = v.x * v.x + v.y * v.y + v.z * v.z + v.w * v.w;
    #pragma unroll
    for (int off = 16; off; off >>= 1) ss += __shfl_xor(ss, off, 32);
    float inv = 1.0f / fmaxf(sqrtf(ss), EPS);
    float4 o = {v.x * inv, v.y * inv, v.z * inv, v.w * inv};
    ((float4*)(out + (size_t)wave * DIM))[lane] = o;
    v4h o16 = {(_Float16)o.x, (_Float16)o.y, (_Float16)o.z, (_Float16)o.w};
    *(v4h*)(out16 + (size_t)wave * DIM + lane * 4) = o16;
}

// ---------------------------------------------------------------------------
extern "C" void kernel_launch(void* const* d_in, const int* in_sizes, int n_in,
                              void* d_out, int out_size, void* d_ws, size_t ws_size,
                              hipStream_t stream) {
    const float* x  = (const float*)d_in[0];
    const int*   ei = (const int*)d_in[1];
    const int N = in_sizes[0] / DIM;
    const int E = in_sizes[1] / 2;
    const int* src = ei;
    const int* dst = ei + E;

    float* out   = (float*)d_out;                   // [4, N, 128]; slice k = h_k
    float* aggr  = (float*)d_ws;                    // N*128 f32
    float* alpha = aggr + (size_t)N * DIM;          // E f32
    size_t h16_off = ((size_t)N * DIM * 4 + (size_t)E * 4 + 255) & ~(size_t)255;
    _Float16* h16 = (_Float16*)((char*)d_ws + h16_off);  // N*128 f16, 256B aligned

    const size_t slice = (size_t)N * DIM;
    dim3 blk(256);                                  // 8 wave32 per block

    // h0 = normalize(x) -> out slice 0 (+ f16 shadow)
    pmwa_norm_kernel<<<dim3((N + 7) / 8), blk, 0, stream>>>(x, out, h16, N);

    const float* h = out;
    for (int k = 0; k < NUM_HOPS; ++k) {
        float* hn = out + (size_t)(k + 1) * slice;

        int waves = (E + 15) / 16;
        pmwa_alpha_wmma_kernel<<<dim3((waves + 7) / 8), blk, 0, stream>>>(
            h16, src, dst, alpha, E);

        hipMemsetAsync(aggr, 0, slice * sizeof(float), stream);

        long long sth = (long long)E * 32;
        pmwa_scatter_kernel<<<dim3((unsigned)((sth + 255) / 256)), blk, 0, stream>>>(
            h, src, dst, alpha, aggr, E);

        pmwa_finalize_kernel<<<dim3((N + 7) / 8), blk, 0, stream>>>(
            aggr, hn, h16, N, k);
        h = hn;
    }
}
